// Spatial_Attention_9423158248224
// MI455X (gfx1250) — matrix operand order repaired and checked by reading, not compiled, not run on hardware
//
#include <hip/hip_runtime.h>
#include <hip/hip_bf16.h>

// Spatial attention on MI455X (gfx1250): all three GEMM stages run on
// v_wmma_f32_16x16x32_bf16 (wave32). Intermediates stored bf16, transposed so
// every WMMA fragment load is contiguous. The dominant stage (k_refine,
// ~174 GFLOP) uses NO LDS and NO barriers: both operands are fragment-loaded
// straight from global (cam/attnT are L2-resident), with fp32->bf16 packing
// done in registers on adjacent pairs (clean v_cvt_pk_bf16_f32).

#define Bb   32
#define CC   2048
#define NN   1152   // 48*24
#define MIDD 128

typedef __attribute__((ext_vector_type(16))) __bf16 v16bf;
typedef __attribute__((ext_vector_type(8)))  float  v8f;

union Frag16  { v16bf v; uint4 u[2]; };
union Pack8   { uint4 u; __bf16 h[8]; };
union PackB64 { uint2 u; __bf16 h[4]; };

__device__ __forceinline__ unsigned pack2(float lo, float hi) {
  union { unsigned u; __bf16 h[2]; } t;
  t.h[0] = (__bf16)lo; t.h[1] = (__bf16)hi;
  return t.u;  // lowers to v_cvt_pk_bf16_f32
}

// Fragment element mapping assumed (per CDNA5 ISA 7.12.2), lane = tid&31,
// half = lane>>4, r = lane&15:
//   A (16x32, M x K):  a[e] = A[r][ half*8 + (e&7) + (e>=8 ? 16 : 0) ]
//   B (32x16, K x N):  b[e] = B[ half*8 + (e&7) + (e>=8 ? 16 : 0) ][ r ]
//   D (16x16 f32):     d[g] -> row M = half*8 + g, col N = r
__device__ __forceinline__ v16bf ld_frag2(const __bf16* p0, const __bf16* p1) {
  Frag16 f;
  f.u[0] = *reinterpret_cast<const uint4*>(p0);
  f.u[1] = *reinterpret_cast<const uint4*>(p1);
  return f.v;
}

// A fragment directly from an fp32 row (K-contiguous): two 8-float runs
// (4x global_load_b128) packed pairwise to bf16.
__device__ __forceinline__ v16bf ld_afrag_f32(const float* row, int kbase) {
  const float4 q0 = *reinterpret_cast<const float4*>(row + kbase);
  const float4 q1 = *reinterpret_cast<const float4*>(row + kbase + 4);
  const float4 q2 = *reinterpret_cast<const float4*>(row + kbase + 16);
  const float4 q3 = *reinterpret_cast<const float4*>(row + kbase + 20);
  Frag16 f;
  f.u[0].x = pack2(q0.x, q0.y); f.u[0].y = pack2(q0.z, q0.w);
  f.u[0].z = pack2(q1.x, q1.y); f.u[0].w = pack2(q1.z, q1.w);
  f.u[1].x = pack2(q2.x, q2.y); f.u[1].y = pack2(q2.z, q2.w);
  f.u[1].z = pack2(q3.x, q3.y); f.u[1].w = pack2(q3.z, q3.w);
  return f.v;
}

__device__ __forceinline__ v8f wmma_bf16(v16bf a, v16bf b, v8f c) {
  return __builtin_amdgcn_wmma_f32_16x16x32_bf16(false, a, false, b,
                                                 (short)0, c, false, false);
}

// ---------------------------------------------------------------------------
// Kernel 1: Qt[b][n][m] = sum_c Wq[m][c] * X[b][c][n]   (and Kt with Wk)
// GEMM M=128(m), N=1152(n), K=2048(c). Output stored transposed [n][m] bf16.
// Block: 256 thr = 8 waves (2x4), tile 64(m) x 128(n), K chunks of 32.
// Double-buffered LDS; packed ds_store_b64 (W) / ds_store_b32 (X transpose).
// ---------------------------------------------------------------------------
__global__ __launch_bounds__(256) void k_project(
    const float* __restrict__ feat, const float* __restrict__ Wq,
    const float* __restrict__ Wk, __bf16* __restrict__ Qt,
    __bf16* __restrict__ Kt) {
  const int nb = blockIdx.x;            // 0..8
  const int mb = blockIdx.y;            // 0..1
  const int bz = blockIdx.z;            // batch*2 + wsel
  const int b = bz >> 1, wsel = bz & 1;
  const float* W = wsel ? Wk : Wq;
  __bf16* Out = (wsel ? Kt : Qt) + (size_t)b * NN * MIDD;
  const float* X = feat + (size_t)b * CC * NN;

  const int tid = threadIdx.x;
  const int lane = tid & 31, wave = tid >> 5;
  const int half = lane >> 4, r = lane & 15;
  const int wm = wave >> 2, wn = wave & 3;
  const int m0 = mb * 64, n0 = nb * 128;

  __shared__ __bf16 At[2][64 * 40];   // [m][k], pitch 40 (80B rows)
  __shared__ __bf16 Bt[2][128 * 40];  // [n][k] (X tile transposed)

  v8f acc[2][2] = {};

  float4 wv[2], xa[2], xb[2];
  int wrow[2], wc4[2], xkp[2], xn4[2];

  auto fetch = [&](int kc) {
#pragma unroll
    for (int s = 0; s < 2; ++s) {
      int idx = tid + s * 256;                        // 0..511
      wrow[s] = idx >> 3; wc4[s] = (idx & 7) << 2;    // W: 64 rows x 8 float4
      wv[s] = *reinterpret_cast<const float4*>(
          W + (size_t)(m0 + wrow[s]) * CC + kc + wc4[s]);
      xkp[s] = idx >> 5; xn4[s] = (idx & 31) << 2;    // X: 16 k-pairs x 32 n4
      const float* px = X + (size_t)(kc + 2 * xkp[s]) * NN + n0 + xn4[s];
      xa[s] = *reinterpret_cast<const float4*>(px);        // row k
      xb[s] = *reinterpret_cast<const float4*>(px + NN);   // row k+1
    }
  };
  auto commit = [&](int buf) {
#pragma unroll
    for (int s = 0; s < 2; ++s) {
      PackB64 p;
      p.h[0] = (__bf16)wv[s].x; p.h[1] = (__bf16)wv[s].y;
      p.h[2] = (__bf16)wv[s].z; p.h[3] = (__bf16)wv[s].w;
      *reinterpret_cast<uint2*>(&At[buf][wrow[s] * 40 + wc4[s]]) = p.u;
      const int kk = 2 * xkp[s], nb4 = xn4[s];
      *reinterpret_cast<unsigned*>(&Bt[buf][(nb4 + 0) * 40 + kk]) =
          pack2(xa[s].x, xb[s].x);
      *reinterpret_cast<unsigned*>(&Bt[buf][(nb4 + 1) * 40 + kk]) =
          pack2(xa[s].y, xb[s].y);
      *reinterpret_cast<unsigned*>(&Bt[buf][(nb4 + 2) * 40 + kk]) =
          pack2(xa[s].z, xb[s].z);
      *reinterpret_cast<unsigned*>(&Bt[buf][(nb4 + 3) * 40 + kk]) =
          pack2(xa[s].w, xb[s].w);
    }
  };

  fetch(0);
  commit(0);
  __syncthreads();

  for (int chunk = 0; chunk < CC / 32; ++chunk) {
    const int buf = chunk & 1;
    const bool has_next = (chunk + 1) < CC / 32;
    if (has_next) fetch((chunk + 1) * 32);

    v16bf a[2], bfr[2];
#pragma unroll
    for (int sm = 0; sm < 2; ++sm) {
      const __bf16* p = &At[buf][(wm * 32 + sm * 16 + r) * 40 + half * 8];
      a[sm] = ld_frag2(p, p + 16);
    }
#pragma unroll
    for (int sn = 0; sn < 2; ++sn) {
      const __bf16* p = &Bt[buf][(wn * 32 + sn * 16 + r) * 40 + half * 8];
      bfr[sn] = ld_frag2(p, p + 16);
    }
#pragma unroll
    for (int sm = 0; sm < 2; ++sm)
#pragma unroll
      for (int sn = 0; sn < 2; ++sn)
        acc[sm][sn] = wmma_bf16(a[sm], bfr[sn], acc[sm][sn]);

    if (has_next) commit(buf ^ 1);
    __syncthreads();
  }

  // Transposed store: lane writes 8 consecutive m values -> one 16B store.
#pragma unroll
  for (int sm = 0; sm < 2; ++sm)
#pragma unroll
    for (int sn = 0; sn < 2; ++sn) {
      Pack8 p;
#pragma unroll
      for (int g = 0; g < 8; ++g) p.h[g] = (__bf16)acc[sm][sn][g];
      int n = n0 + wn * 32 + sn * 16 + r;
      int m = m0 + wm * 32 + sm * 16 + half * 8;
      *reinterpret_cast<uint4*>(Out + (size_t)n * MIDD + m) = p.u;
    }
}

// ---------------------------------------------------------------------------
// Kernel 2: S[i][j] = sum_m Qt[i][m]*Kt[j][m]; fused row softmax; store
// attnT[j][i] bf16. Block: 16 rows x all 1152 cols; wave w owns 144 cols
// (9 frags); K=128 in 4 chunks. Fragments loaded straight from global
// (Qt/Kt are K-contiguous; ~19MB total, L2-resident).
// ---------------------------------------------------------------------------
__global__ __launch_bounds__(256) void k_attn(
    const __bf16* __restrict__ Qt, const __bf16* __restrict__ Kt,
    __bf16* __restrict__ attnT) {
  const int i0 = blockIdx.x * 16;
  const int b = blockIdx.y;
  const __bf16* Q = Qt + (size_t)b * NN * MIDD;
  const __bf16* K = Kt + (size_t)b * NN * MIDD;
  __bf16* A = attnT + (size_t)b * NN * NN;

  const int tid = threadIdx.x, lane = tid & 31, wave = tid >> 5;
  const int half = lane >> 4, r = lane & 15;
  const int jbase = wave * 144;

  __shared__ float red[16][8];
  __shared__ float rowv[16];

  v8f acc[9] = {};

#pragma unroll
  for (int kc = 0; kc < MIDD; kc += 32) {
    const __bf16* pa = Q + (size_t)(i0 + r) * MIDD + kc + half * 8;
    v16bf a = ld_frag2(pa, pa + 16);
#pragma unroll
    for (int f = 0; f < 9; ++f) {
      int j = jbase + f * 16 + r;
      const __bf16* pb = K + (size_t)j * MIDD + kc + half * 8;
      v16bf bf = ld_frag2(pb, pb + 16);
      acc[f] = wmma_bf16(a, bf, acc[f]);
    }
  }

  // --- softmax over j (1152 per row), rows = i0 + half*8 + g ---
#pragma unroll
  for (int g = 0; g < 8; ++g) {
    float m = acc[0][g];
#pragma unroll
    for (int f = 1; f < 9; ++f) m = fmaxf(m, acc[f][g]);
    for (int s = 1; s < 16; s <<= 1) m = fmaxf(m, __shfl_xor(m, s, 32));
    if (r == 0) red[half * 8 + g][wave] = m;
  }
  __syncthreads();
  if (tid < 16) {
    float m = red[tid][0];
    for (int w = 1; w < 8; ++w) m = fmaxf(m, red[tid][w]);
    rowv[tid] = m;
  }
  __syncthreads();
#pragma unroll
  for (int g = 0; g < 8; ++g) {
    float rm = rowv[half * 8 + g];
    float s = 0.f;
#pragma unroll
    for (int f = 0; f < 9; ++f) {
      float e = __expf(acc[f][g] - rm);
      acc[f][g] = e;
      s += e;
    }
    for (int t = 1; t < 16; t <<= 1) s += __shfl_xor(s, t, 32);
    if (r == 0) red[half * 8 + g][wave] = s;
  }
  __syncthreads();
  if (tid < 16) {
    float s = red[tid][0];
    for (int w = 1; w < 8; ++w) s += red[tid][w];
    rowv[tid] = 1.0f / s;
  }
  __syncthreads();

  // normalize + transposed bf16 store: one 16B store per fragment
#pragma unroll
  for (int f = 0; f < 9; ++f) {
    Pack8 p;
#pragma unroll
    for (int g = 0; g < 8; ++g)
      p.h[g] = (__bf16)(acc[f][g] * rowv[half * 8 + g]);
    int j = jbase + f * 16 + r;
    *reinterpret_cast<uint4*>(A + (size_t)j * NN + i0 + half * 8) = p.u;
  }
}

// ---------------------------------------------------------------------------
// Kernel 3: out[c][j] = alpha * sum_i cam[c][i]*attn[i][j] + cam[c][j]
// GEMM M=2048(c), N=1152(j), K=1152(i). Dominant ~174 GFLOP stage.
// NO LDS, NO barriers: A fragments read directly from fp32 cam rows
// (4x b128 + paired cvt_pk), B fragments directly from bf16 attnT rows
// (2x b128). Both operands are L2-resident per batch (9.4MB + 2.6MB).
// Wave tile 32(c) x 64(j) = 8 WMMA per 32-K chunk; block 128x128.
// ---------------------------------------------------------------------------
__global__ __launch_bounds__(256) void k_refine(
    const float* __restrict__ cam, const __bf16* __restrict__ attnT,
    const float* __restrict__ alphaPtr, float* __restrict__ out) {
  const int jb = blockIdx.x;     // 0..8   (j tiles of 128)
  const int cb = blockIdx.y;     // 0..15  (c tiles of 128)
  const int b = blockIdx.z;
  const float* Cam = cam + (size_t)b * CC * NN;
  const __bf16* A = attnT + (size_t)b * NN * NN;
  float* O = out + (size_t)b * CC * NN;
  const float alpha = alphaPtr[0];

  const int tid = threadIdx.x, lane = tid & 31, wave = tid >> 5;
  const int half = lane >> 4, r = lane & 15;
  const int wc = wave >> 1, wj = wave & 1;          // 4 x 2 waves
  const int cRow0 = cb * 128 + wc * 32;             // wave's c base
  const int jRow0 = jb * 128 + wj * 64;             // wave's j base

  v8f acc[2][4] = {};

  for (int kc = 0; kc < NN; kc += 32) {
    v16bf a[2], bfr[4];
#pragma unroll
    for (int sm = 0; sm < 2; ++sm)
      a[sm] = ld_afrag_f32(Cam + (size_t)(cRow0 + sm * 16 + r) * NN,
                           kc + half * 8);
#pragma unroll
    for (int sn = 0; sn < 4; ++sn) {
      const __bf16* p = A + (size_t)(jRow0 + sn * 16 + r) * NN + kc + half * 8;
      bfr[sn] = ld_frag2(p, p + 16);
    }
#pragma unroll
    for (int sm = 0; sm < 2; ++sm)
#pragma unroll
      for (int sn = 0; sn < 4; ++sn)
        acc[sm][sn] = wmma_bf16(a[sm], bfr[sn], acc[sm][sn]);
  }

#pragma unroll
  for (int sm = 0; sm < 2; ++sm)
#pragma unroll
    for (int sn = 0; sn < 4; ++sn) {
      int c = cRow0 + sm * 16 + half * 8;
      int j = jRow0 + sn * 16 + r;
#pragma unroll
      for (int g = 0; g < 8; ++g) {
        size_t idx = (size_t)(c + g) * NN + j;
        O[idx] = alpha * acc[sm][sn][g] + Cam[idx];
      }
    }
}

// ---------------------------------------------------------------------------
extern "C" void kernel_launch(void* const* d_in, const int* in_sizes, int n_in,
                              void* d_out, int out_size, void* d_ws,
                              size_t ws_size, hipStream_t stream) {
  const float* feat  = (const float*)d_in[0];
  const float* cam   = (const float*)d_in[1];
  const float* Wq    = (const float*)d_in[2];
  const float* Wk    = (const float*)d_in[3];
  const float* alpha = (const float*)d_in[4];
  float* out = (float*)d_out;

  // Workspace: Qt, Kt = [B][N][MID] bf16; attnT = [B][N][N] bf16 (~99 MB).
  const size_t QK = (size_t)Bb * NN * MIDD;       // elements
  __bf16* Qt = (__bf16*)d_ws;
  __bf16* Kt = Qt + QK;
  __bf16* At = Kt + QK;

  k_project<<<dim3(9, 2, Bb * 2), 256, 0, stream>>>(feat, Wq, Wk, Qt, Kt);
  k_attn<<<dim3(NN / 16, Bb), 256, 0, stream>>>(Qt, Kt, At);
  k_refine<<<dim3(9, CC / 128, Bb), 256, 0, stream>>>(cam, At, alpha, out);
}
